// ResonantColorMatcher_22393959481785
// MI455X (gfx1250) — matrix operand order repaired and checked by reading, not compiled, not run on hardware
//
#include <hip/hip_runtime.h>

typedef __bf16 bf16x16 __attribute__((ext_vector_type(16)));
typedef float  f32x8   __attribute__((ext_vector_type(8)));

union FragB {
    bf16x16 v;
    unsigned int u[8];
    unsigned short h[16];
};

#define N_ 4096
#define C_ 64
#define B_ 2
#define DT_ 0.2f
#define RW_ 0.15f
#define SCALE_ 0.125f   // 64^-0.5

__device__ __forceinline__ unsigned short f2bf(float x) {
    unsigned u = __float_as_uint(x);
    unsigned r = u + 0x7FFFu + ((u >> 16) & 1u);   // round-to-nearest-even
    return (unsigned short)(r >> 16);
}

__device__ __forceinline__ f32x8 wmma_bf16(bf16x16 a, bf16x16 b, f32x8 c) {
    return __builtin_amdgcn_wmma_f32_16x16x32_bf16(false, a, false, b, (short)0, c, false, false);
}

// CDNA5 async DMA: copy 16B global -> LDS, tracked by ASYNCcnt (no VGPR staging)
__device__ __forceinline__ void async_copy16(unsigned lds_off, const void* gaddr) {
    asm volatile("global_load_async_to_lds_b128 %0, %1, off"
                 :: "v"(lds_off), "v"(gaddr)
                 : "memory");
}
__device__ __forceinline__ void wait_async0() {
    asm volatile("s_wait_asynccnt 0" ::: "memory");
}

// A-fragment (16x32, M=rows, K=channels): lane<16 row=base+l, K {0..7,16..23}(+32*chalf)
//                                         lane>=16          K {8..15,24..31}
__device__ __forceinline__ bf16x16 loadA_frag(const unsigned short* row, int chalf, int lane) {
    FragB f;
    const int off = chalf * 32 + ((lane & 16) ? 8 : 0);
    uint4 a = *(const uint4*)(row + off);
    uint4 b = *(const uint4*)(row + off + 16);
    f.u[0] = a.x; f.u[1] = a.y; f.u[2] = a.z; f.u[3] = a.w;
    f.u[4] = b.x; f.u[5] = b.y; f.u[6] = b.z; f.u[7] = b.w;
    return f.v;
}

// B-fragment (32x16, K=rows, N=cols): lane<16 col=base+l, K {0..7,16..23}(+32*chalf)
//                                     lane>=16            K {8..15,24..31}
__device__ __forceinline__ bf16x16 loadB_frag(const unsigned short* row, int chalf, int lane) {
    FragB f;
    const int off = chalf * 32 + ((lane & 16) ? 8 : 0);
    uint4 a = *(const uint4*)(row + off);
    uint4 b = *(const uint4*)(row + off + 16);
    f.u[0] = a.x; f.u[1] = a.y; f.u[2] = a.z; f.u[3] = a.w;
    f.u[4] = b.x; f.u[5] = b.y; f.u[6] = b.z; f.u[7] = b.w;
    return f.v;
}

// 16 bf16: p[0..7], p[16..23] (V B-fragment reads)
__device__ __forceinline__ bf16x16 load16_frag(const unsigned short* p) {
    FragB f;
    uint4 a = *(const uint4*)(p);
    uint4 b = *(const uint4*)(p + 16);
    f.u[0] = a.x; f.u[1] = a.y; f.u[2] = a.z; f.u[3] = a.w;
    f.u[4] = b.x; f.u[5] = b.y; f.u[6] = b.z; f.u[7] = b.w;
    return f.v;
}

// ---------------------------------------------------------------- avgpool 4x4
__global__ __launch_bounds__(256) void avgpool4_k(const float* __restrict__ in,
                                                  float* __restrict__ out) {
    int idx = blockIdx.x * 256 + threadIdx.x;          // [B][C][64][64]
    int ox = idx & 63, oy = (idx >> 6) & 63, c = (idx >> 12) & 63, b = idx >> 18;
    const float* p = in + (((size_t)(b * 64 + c) * 256 + oy * 4) * 256 + ox * 4);
    float s = 0.f;
#pragma unroll
    for (int j = 0; j < 4; ++j)
#pragma unroll
        for (int i = 0; i < 4; ++i) s += p[j * 256 + i];
    out[idx] = s * (1.0f / 16.0f);
}

// ----------------------------------------------------- conv1x1 with activation
// in [B][C][N], w [O][C], out [B][O][N]; act: 0 none, 1 2*sigmoid, 2 sigmoid
__global__ __launch_bounds__(256) void conv1x1_k(const float* __restrict__ in,
                                                 const float* __restrict__ w,
                                                 float* __restrict__ out, int act) {
    int idx = blockIdx.x * 256 + threadIdx.x;
    int n = idx & (N_ - 1), o = (idx >> 12) & 63, b = idx >> 18;
    const float* ip = in + (size_t)b * (C_ * N_) + n;
    const float* wp = w + o * 64;
    float acc = 0.f;
#pragma unroll 8
    for (int c = 0; c < 64; ++c) acc = fmaf(wp[c], ip[(size_t)c * N_], acc);
    if (act == 1) acc = 2.0f / (1.0f + __expf(-acc));
    else if (act == 2) acc = 1.0f / (1.0f + __expf(-acc));
    out[idx] = acc;
}

// ----------------------------------------------------------- evolve: first step
__global__ __launch_bounds__(256) void init_xv_k(const float* __restrict__ force,
                                                 float* __restrict__ x, float* __restrict__ v) {
    int idx = blockIdx.x * 256 + threadIdx.x;
    float f = force[idx];
    v[idx] = DT_ * f;
    x[idx] = DT_ * DT_ * f;
}

// ------------------------------------------------- evolve: symplectic Euler step
__global__ __launch_bounds__(256) void evolve_step_k(
    const float* __restrict__ x, const float* __restrict__ v,
    const float* __restrict__ force, const float* __restrict__ omega,
    const float* __restrict__ zeta, const float* __restrict__ wdw,
    const float* __restrict__ wpw, float* __restrict__ xn, float* __restrict__ vn) {
    int idx = blockIdx.x * 256 + threadIdx.x;          // [B][C][64][64]
    int xc = idx & 63, y = (idx >> 6) & 63, o = (idx >> 12) & 63, b = idx >> 18;
    const float* xo = x + (size_t)(b * 64 + o) * 4096;
    float acc = 0.f;
#pragma unroll
    for (int ky = -1; ky <= 1; ++ky) {
        int yy = y + ky;
        if (yy < 0 || yy > 63) continue;
#pragma unroll
        for (int kx = -1; kx <= 1; ++kx) {
            int xx = xc + kx;
            if (xx < 0 || xx > 63) continue;
            acc = fmaf(wdw[o * 9 + (ky + 1) * 3 + (kx + 1)], xo[yy * 64 + xx], acc);
        }
    }
    const float* xb = x + (size_t)b * (C_ * N_) + y * 64 + xc;
    const float* wp = wpw + o * 64;
#pragma unroll 8
    for (int c = 0; c < 64; ++c) acc = fmaf(wp[c], xb[(size_t)c * N_], acc);
    float f = force[idx], om = omega[idx], ze = zeta[idx];
    float xv = x[idx], vv = v[idx];
    float a = f + acc - 2.0f * ze * om * vv - om * om * xv;
    float v2 = vv + a * DT_;
    vn[idx] = v2;
    xn[idx] = xv + v2 * DT_;
}

// ----------------------------------------- transpose+pack fp32 [B][C][N] -> bf16 [B][N][C]
__global__ __launch_bounds__(256) void transpose_pack_k(const float* __restrict__ in,
                                                        unsigned short* __restrict__ out) {
    __shared__ float t[16][17];
    int b = blockIdx.z, c0 = blockIdx.y * 16, n0 = blockIdx.x * 16;
    int tx = threadIdx.x, ty = threadIdx.y;
    t[ty][tx] = in[((size_t)b * 64 + c0 + ty) * N_ + n0 + tx];
    __syncthreads();
    out[((size_t)b * N_ + n0 + ty) * 64 + c0 + tx] = f2bf(t[tx][ty]);
}

// ---------------------------------------------------- elementwise fp32 -> bf16
__global__ __launch_bounds__(256) void convert_bf_k(const float* __restrict__ in,
                                                    unsigned short* __restrict__ out) {
    int idx = blockIdx.x * 256 + threadIdx.x;
    out[idx] = f2bf(in[idx]);
}

// ------------------------------------------- squared row norms over channels (fp32)
__global__ __launch_bounds__(256) void rownorm_k(const float* __restrict__ in,
                                                 float* __restrict__ out) {
    int i = blockIdx.x * 256 + threadIdx.x;            // b*N + n
    int b = i >> 12, n = i & (N_ - 1);
    const float* p = in + (size_t)b * (C_ * N_) + n;
    float s = 0.f;
#pragma unroll 8
    for (int c = 0; c < 64; ++c) { float v = p[(size_t)c * N_]; s = fmaf(v, v, s); }
    out[i] = s;
}

// =================== fused resonant flash attention (WMMA bf16) ===================
// LDS staging buffer layout (unsigned short indices, per buffer = 8192 shorts = 16KB):
//   [0,2048)    Kf block: 32 rows x 64 ch
//   [2048,4096) Kw block
//   [4096,6144) Kz block
//   [6144,8192) V block: 64 ch-rows x 32 m
#define STG_K  2048
#define STG_V  6144

__device__ __forceinline__ void issue_stage(
    const unsigned short* __restrict__ Kf, const unsigned short* __restrict__ Kw,
    const unsigned short* __restrict__ Kz, const unsigned short* __restrict__ Vt,
    int b, int mb, unsigned short* bufPtr, int t) {
    // 768 K-chunks of 16B (8 bf16): mat = ch>>8, row = (ch&255)>>3, off8 = ch&7
#pragma unroll
    for (int i = 0; i < 6; ++i) {
        int ch = t + i * 128;
        int mat = ch >> 8;
        int rem = ch & 255;
        int row = rem >> 3, off = (rem & 7) * 8;
        const unsigned short* g = (mat == 0 ? Kf : (mat == 1 ? Kw : Kz));
        g += ((size_t)(b * N_ + mb + row)) * 64 + off;
        unsigned lds = (unsigned)(size_t)(bufPtr + mat * STG_K + row * 64 + off);
        async_copy16(lds, g);
    }
    // 256 V-chunks: c = ch>>2, off8 = ch&3
#pragma unroll
    for (int i = 0; i < 2; ++i) {
        int ch = t + i * 128;
        int c = ch >> 2, off = (ch & 3) * 8;
        const unsigned short* g = Vt + ((size_t)(b * 64 + c)) * N_ + mb + off;
        unsigned lds = (unsigned)(size_t)(bufPtr + STG_V + c * 32 + off);
        async_copy16(lds, g);
    }
}

__global__ __launch_bounds__(128) void attn_k(
    const unsigned short* __restrict__ Qf, const unsigned short* __restrict__ Kf,
    const unsigned short* __restrict__ Qw, const unsigned short* __restrict__ Kw,
    const unsigned short* __restrict__ Qz, const unsigned short* __restrict__ Kz,
    const unsigned short* __restrict__ Vt,
    const float* __restrict__ nQw, const float* __restrict__ nQz,
    const float* __restrict__ nKw, const float* __restrict__ nKz,
    float* __restrict__ Ctx) {
    const int b = blockIdx.z;
    const int t = threadIdx.x;
    const int wave = t >> 5;
    const int lane = t & 31;
    const int hi = (lane & 16) ? 1 : 0;
    const int l15 = lane & 15;
    const int n0 = (blockIdx.x * 4 + wave) * 16;
    const int myn = n0 + l15;

    __shared__ float s_nkw[N_];          // 16KB
    __shared__ float s_nkz[N_];          // 16KB
    __shared__ unsigned short stage[2][8192];  // 2 x 16KB double buffer
    for (int i = t; i < N_; i += 128) {
        s_nkw[i] = nKw[b * N_ + i];
        s_nkz[i] = nKz[b * N_ + i];
    }

    // Q fragments (rows fixed for whole kernel), from global
    const unsigned short* qfrow = Qf + ((size_t)b * N_ + myn) * 64;
    const unsigned short* qwrow = Qw + ((size_t)b * N_ + myn) * 64;
    const unsigned short* qzrow = Qz + ((size_t)b * N_ + myn) * 64;
    const bf16x16 bqf0 = loadB_frag(qfrow, 0, lane), bqf1 = loadB_frag(qfrow, 1, lane);
    const bf16x16 bqw0 = loadB_frag(qwrow, 0, lane), bqw1 = loadB_frag(qwrow, 1, lane);
    const bf16x16 bqz0 = loadB_frag(qzrow, 0, lane), bqz1 = loadB_frag(qzrow, 1, lane);
    const float nqw = nQw[b * N_ + myn];
    const float nqz = nQz[b * N_ + myn];

    // prologue: stage first K/V block asynchronously
    issue_stage(Kf, Kw, Kz, Vt, b, 0, &stage[0][0], t);
    wait_async0();
    __syncthreads();

    float M = -1e30f, L = 0.0f;
    f32x8 O0 = {}, O1 = {}, O2 = {}, O3 = {};
    int buf = 0;

    for (int mb = 0; mb < N_; mb += 32) {
        const unsigned short* S = &stage[buf][0];
        if (mb + 32 < N_)   // overlap next block's DMA with this block's WMMA
            issue_stage(Kf, Kw, Kz, Vt, b, mb + 32, &stage[buf ^ 1][0], t);

        float p0[8], p1[8];
        float M1 = M, sum0 = 0.f, sum1 = 0.f;
        const float Mold = M;
#pragma unroll
        for (int sub = 0; sub < 2; ++sub) {
            const int m0 = mb + sub * 16;
            const int lrow = sub * 16 + l15;          // row within 32-row LDS block
            const unsigned short* kfrow = S + 0 * STG_K + lrow * 64;
            const unsigned short* kwrow = S + 1 * STG_K + lrow * 64;
            const unsigned short* kzrow = S + 2 * STG_K + lrow * 64;
            // S^T tile: D[M=m, N=n]; lane holds n=myn, VGPR r holds m=m0+r+8*hi
            f32x8 sf = {}, sw = {}, sz = {};
            {
                bf16x16 a0 = loadA_frag(kfrow, 0, lane), a1 = loadA_frag(kfrow, 1, lane);
                sf = wmma_bf16(a0, bqf0, sf);
                sf = wmma_bf16(a1, bqf1, sf);
            }
            {
                bf16x16 a0 = loadA_frag(kwrow, 0, lane), a1 = loadA_frag(kwrow, 1, lane);
                sw = wmma_bf16(a0, bqw0, sw);
                sw = wmma_bf16(a1, bqw1, sw);
            }
            {
                bf16x16 a0 = loadA_frag(kzrow, 0, lane), a1 = loadA_frag(kzrow, 1, lane);
                sz = wmma_bf16(a0, bqz0, sz);
                sz = wmma_bf16(a1, bqz1, sz);
            }
            float smax = -1e30f;
            float sv[8];
#pragma unroll
            for (int r = 0; r < 8; ++r) {
                const int m = m0 + r + hi * 8;
                float dw2 = fmaxf(nqw + s_nkw[m] - 2.0f * sw[r], 0.0f);
                float dz2 = fmaxf(nqz + s_nkz[m] - 2.0f * sz[r], 0.0f);
                float s = (sf[r] - RW_ * (sqrtf(dw2) + sqrtf(dz2))) * SCALE_;
                sv[r] = s;
                smax = fmaxf(smax, s);
            }
            smax = fmaxf(smax, __shfl_xor(smax, 16, 32));  // combine row halves
            const float Mn = fmaxf(M, smax);
            float rs = 0.f;
#pragma unroll
            for (int r = 0; r < 8; ++r) {
                float p = __expf(sv[r] - Mn);
                rs += p;
                if (sub == 0) p0[r] = p; else p1[r] = p;
            }
            rs += __shfl_xor(rs, 16, 32);
            if (sub == 0) { sum0 = rs; M1 = Mn; } else { sum1 = rs; }
            M = Mn;
        }
        const float c0 = __expf(M1 - M);       // correct sub0 probs to final max
        const float alpha = __expf(Mold - M);  // rescale running state
        L = L * alpha + sum0 * c0 + sum1;

        // P (bf16 A-fragment for PV, K=32): lane<16 row n, m {0..7,16..23};
        // exactly the S^T D-layout -> no cross-lane movement needed.
        FragB pf;
#pragma unroll
        for (int r = 0; r < 8; ++r) {
            pf.h[r] = f2bf(p0[r] * c0);
            pf.h[8 + r] = f2bf(p1[r]);
        }
        // O rows live at VGPR index -> broadcast per-row alpha from owning lanes
#pragma unroll
        for (int r = 0; r < 8; ++r) {
            const float f = __shfl(alpha, r + hi * 8, 32);
            O0[r] *= f; O1[r] *= f; O2[r] *= f; O3[r] *= f;
        }
        const unsigned short* vb = S + STG_V + hi * 8;
        O0 = wmma_bf16(pf.v, load16_frag(vb + (0 * 16 + l15) * 32), O0);
        O1 = wmma_bf16(pf.v, load16_frag(vb + (1 * 16 + l15) * 32), O1);
        O2 = wmma_bf16(pf.v, load16_frag(vb + (2 * 16 + l15) * 32), O2);
        O3 = wmma_bf16(pf.v, load16_frag(vb + (3 * 16 + l15) * 32), O3);

        wait_async0();      // our next-buffer DMA complete
        __syncthreads();    // everyone done reading current buffer
        buf ^= 1;
    }

    const float invL = 1.0f / L;
#pragma unroll
    for (int r = 0; r < 8; ++r) {
        const float f = __shfl(invL, r + hi * 8, 32);
        const int n = n0 + r + hi * 8;
        float* op = Ctx + ((size_t)b * N_ + n) * 64 + l15;
        op[0]  = O0[r] * f;
        op[16] = O1[r] * f;
        op[32] = O2[r] * f;
        op[48] = O3[r] * f;
    }
}

// ------------------------------- w_out conv: ctx [B][N][C] -> ctx2 [B][C][N]
__global__ __launch_bounds__(256) void conv_out_k(const float* __restrict__ ctx,
                                                  const float* __restrict__ w,
                                                  float* __restrict__ out) {
    int idx = blockIdx.x * 256 + threadIdx.x;
    int n = idx & (N_ - 1), o = (idx >> 12) & 63, b = idx >> 18;
    const float* ip = ctx + ((size_t)b * N_ + n) * 64;
    const float* wp = w + o * 64;
    float acc = 0.f;
#pragma unroll 8
    for (int c = 0; c < 64; ++c) acc = fmaf(wp[c], ip[c], acc);
    out[(size_t)b * (C_ * N_) + (size_t)o * N_ + n] = acc;
}

// -------------------- fused bilinear x4 upsample + LeakyReLU gate + residual
__global__ __launch_bounds__(256) void gate_k(const float* __restrict__ src,
                                              const float* __restrict__ ctx2,
                                              const float* __restrict__ wg1,
                                              const float* __restrict__ wg2,
                                              float* __restrict__ out) {
    __shared__ float in_ld[64][130];   // [pixel][128 channels], padded
    __shared__ float g1_ld[64][66];
    const int b = blockIdx.z, y = blockIdx.y, x0 = blockIdx.x * 64;
    const int t = threadIdx.x, p = t & 63, grp = t >> 6;
    const int x = x0 + p;

    float sy = (y + 0.5f) * 0.25f - 0.5f;
    float sx = (x + 0.5f) * 0.25f - 0.5f;
    int yi = (int)floorf(sy), xi = (int)floorf(sx);
    float wy1 = sy - yi, wy0 = 1.0f - wy1;
    float wx1 = sx - xi, wx0 = 1.0f - wx1;
    int ya = min(max(yi, 0), 63), yb = min(max(yi + 1, 0), 63);
    int xa = min(max(xi, 0), 63), xb = min(max(xi + 1, 0), 63);

    for (int cc = grp * 32; cc < grp * 32 + 32; ++cc) {
        float val;
        if (cc < 64) {
            val = src[(((size_t)b * 64 + cc) * 256 + y) * 256 + x];
        } else {
            const float* cp = ctx2 + (size_t)(b * 64 + (cc - 64)) * 4096;
            val = wy0 * (wx0 * cp[ya * 64 + xa] + wx1 * cp[ya * 64 + xb]) +
                  wy1 * (wx0 * cp[yb * 64 + xa] + wx1 * cp[yb * 64 + xb]);
        }
        in_ld[p][cc] = val;
    }
    __syncthreads();

    for (int o = grp * 16; o < grp * 16 + 16; ++o) {
        float acc = 0.f;
        const float* wp = wg1 + o * 128;
#pragma unroll 8
        for (int c = 0; c < 128; ++c) acc = fmaf(wp[c], in_ld[p][c], acc);
        g1_ld[p][o] = acc > 0.f ? acc : 0.2f * acc;
    }
    __syncthreads();

    for (int o = grp * 16; o < grp * 16 + 16; ++o) {
        float acc = 0.f;
        const float* wp = wg2 + o * 64;
#pragma unroll 8
        for (int c = 0; c < 64; ++c) acc = fmaf(wp[c], g1_ld[p][c], acc);
        float g = 1.0f / (1.0f + __expf(-acc));
        out[(((size_t)b * 64 + o) * 256 + y) * 256 + x] = in_ld[p][o] + g * in_ld[p][64 + o];
    }
}

// =============================================================================
extern "C" void kernel_launch(void* const* d_in, const int* in_sizes, int n_in,
                              void* d_out, int out_size, void* d_ws, size_t ws_size,
                              hipStream_t stream) {
    const float* src      = (const float*)d_in[0];
    const float* ref      = (const float*)d_in[1];
    const float* w_q_in   = (const float*)d_in[2];
    const float* w_k_in   = (const float*)d_in[3];
    const float* w_v_in   = (const float*)d_in[4];
    const float* w_q_dw   = (const float*)d_in[5];
    const float* w_q_pw   = (const float*)d_in[6];
    const float* w_k_dw   = (const float*)d_in[7];
    const float* w_k_pw   = (const float*)d_in[8];
    const float* w_v_dw   = (const float*)d_in[9];
    const float* w_v_pw   = (const float*)d_in[10];
    const float* w_omega_q = (const float*)d_in[11];
    const float* w_zeta_q  = (const float*)d_in[12];
    const float* w_omega_k = (const float*)d_in[13];
    const float* w_zeta_k  = (const float*)d_in[14];
    const float* w_omega_v = (const float*)d_in[15];
    const float* w_zeta_v  = (const float*)d_in[16];
    const float* w_out    = (const float*)d_in[17];
    const float* w_gate1  = (const float*)d_in[18];
    const float* w_gate2  = (const float*)d_in[19];
    float* out = (float*)d_out;

    const size_t SZ = (size_t)B_ * C_ * N_;      // 524288 elements
    char* p = (char*)d_ws;
    auto alloc = [&](size_t bytes) -> void* {
        void* r = (void*)p;
        p += (bytes + 255) & ~(size_t)255;
        return r;
    };
    float* s_pool = (float*)alloc(SZ * 4);
    float* r_pool = (float*)alloc(SZ * 4);
    float* force  = (float*)alloc(SZ * 4);
    float* om_q = (float*)alloc(SZ * 4); float* ze_q = (float*)alloc(SZ * 4);
    float* om_k = (float*)alloc(SZ * 4); float* ze_k = (float*)alloc(SZ * 4);
    float* om_v = (float*)alloc(SZ * 4); float* ze_v = (float*)alloc(SZ * 4);
    float* qx = (float*)alloc(SZ * 4);
    float* kx = (float*)alloc(SZ * 4);
    float* vx = (float*)alloc(SZ * 4);
    float* vA = (float*)alloc(SZ * 4);
    float* xB = (float*)alloc(SZ * 4);
    float* vB = (float*)alloc(SZ * 4);
    float* ctx  = (float*)alloc(SZ * 4);
    float* ctx2 = (float*)alloc(SZ * 4);
    unsigned short* qf = (unsigned short*)alloc(SZ * 2);
    unsigned short* kf = (unsigned short*)alloc(SZ * 2);
    unsigned short* qw = (unsigned short*)alloc(SZ * 2);
    unsigned short* kw = (unsigned short*)alloc(SZ * 2);
    unsigned short* qz = (unsigned short*)alloc(SZ * 2);
    unsigned short* kz = (unsigned short*)alloc(SZ * 2);
    unsigned short* vt = (unsigned short*)alloc(SZ * 2);
    float* nqw = (float*)alloc((size_t)B_ * N_ * 4);
    float* nqz = (float*)alloc((size_t)B_ * N_ * 4);
    float* nkw = (float*)alloc((size_t)B_ * N_ * 4);
    float* nkz = (float*)alloc((size_t)B_ * N_ * 4);

    const int EB = (int)(SZ / 256);   // 2048 blocks for elementwise/conv grids

    avgpool4_k<<<EB, 256, 0, stream>>>(src, s_pool);
    avgpool4_k<<<EB, 256, 0, stream>>>(ref, r_pool);

    // --- evolve q (from s_pool) ---
    conv1x1_k<<<EB, 256, 0, stream>>>(s_pool, w_omega_q, om_q, 1);
    conv1x1_k<<<EB, 256, 0, stream>>>(s_pool, w_zeta_q, ze_q, 2);
    conv1x1_k<<<EB, 256, 0, stream>>>(s_pool, w_q_in, force, 0);
    init_xv_k<<<EB, 256, 0, stream>>>(force, qx, vA);
    evolve_step_k<<<EB, 256, 0, stream>>>(qx, vA, force, om_q, ze_q, w_q_dw, w_q_pw, xB, vB);
    evolve_step_k<<<EB, 256, 0, stream>>>(xB, vB, force, om_q, ze_q, w_q_dw, w_q_pw, qx, vA);

    // --- evolve k (from r_pool) ---
    conv1x1_k<<<EB, 256, 0, stream>>>(r_pool, w_omega_k, om_k, 1);
    conv1x1_k<<<EB, 256, 0, stream>>>(r_pool, w_zeta_k, ze_k, 2);
    conv1x1_k<<<EB, 256, 0, stream>>>(r_pool, w_k_in, force, 0);
    init_xv_k<<<EB, 256, 0, stream>>>(force, kx, vA);
    evolve_step_k<<<EB, 256, 0, stream>>>(kx, vA, force, om_k, ze_k, w_k_dw, w_k_pw, xB, vB);
    evolve_step_k<<<EB, 256, 0, stream>>>(xB, vB, force, om_k, ze_k, w_k_dw, w_k_pw, kx, vA);

    // --- evolve v (from r_pool) ---
    conv1x1_k<<<EB, 256, 0, stream>>>(r_pool, w_omega_v, om_v, 1);
    conv1x1_k<<<EB, 256, 0, stream>>>(r_pool, w_zeta_v, ze_v, 2);
    conv1x1_k<<<EB, 256, 0, stream>>>(r_pool, w_v_in, force, 0);
    init_xv_k<<<EB, 256, 0, stream>>>(force, vx, vA);
    evolve_step_k<<<EB, 256, 0, stream>>>(vx, vA, force, om_v, ze_v, w_v_dw, w_v_pw, xB, vB);
    evolve_step_k<<<EB, 256, 0, stream>>>(xB, vB, force, om_v, ze_v, w_v_dw, w_v_pw, vx, vA);

    // --- pack to bf16 layouts + norms ---
    dim3 tg(N_ / 16, C_ / 16, B_), tb(16, 16);
    transpose_pack_k<<<tg, tb, 0, stream>>>(qx, qf);
    transpose_pack_k<<<tg, tb, 0, stream>>>(kx, kf);
    transpose_pack_k<<<tg, tb, 0, stream>>>(om_q, qw);
    transpose_pack_k<<<tg, tb, 0, stream>>>(om_k, kw);
    transpose_pack_k<<<tg, tb, 0, stream>>>(ze_q, qz);
    transpose_pack_k<<<tg, tb, 0, stream>>>(ze_k, kz);
    convert_bf_k<<<EB, 256, 0, stream>>>(vx, vt);
    rownorm_k<<<(B_ * N_) / 256, 256, 0, stream>>>(om_q, nqw);
    rownorm_k<<<(B_ * N_) / 256, 256, 0, stream>>>(ze_q, nqz);
    rownorm_k<<<(B_ * N_) / 256, 256, 0, stream>>>(om_k, nkw);
    rownorm_k<<<(B_ * N_) / 256, 256, 0, stream>>>(ze_k, nkz);

    // --- fused resonant flash attention (WMMA + async LDS staging) ---
    attn_k<<<dim3(N_ / 64, 1, B_), 128, 0, stream>>>(qf, kf, qw, kw, qz, kz, vt,
                                                     nqw, nqz, nkw, nkz, ctx);

    // --- output projection + fused upsample/gate ---
    conv_out_k<<<EB, 256, 0, stream>>>(ctx, w_out, ctx2);
    gate_k<<<dim3(4, 256, B_), 256, 0, stream>>>(src, ctx2, w_gate1, w_gate2, out);

    (void)in_sizes; (void)n_in; (void)out_size; (void)ws_size;
}
